// GATv2_29094108463830
// MI455X (gfx1250) — compile-verified, matrix operand order checked by reading, not run on hardware
//
#include <hip/hip_runtime.h>
#include <hip/hip_bf16.h>
#include <math.h>

#define D 128
#define SLOPE 0.2f
#define LDSW 136  // padded row stride (halves) to avoid 16-way LDS bank conflicts

typedef __attribute__((ext_vector_type(16))) _Float16 v16h;
typedef __attribute__((ext_vector_type(8)))  _Float16 h8;
typedef __attribute__((ext_vector_type(8)))  float    v8f;

// ---------------------------------------------------------------------------
// helpers
// ---------------------------------------------------------------------------
__device__ __forceinline__ void atomicMaxF(float* addr, float v) {
    // IEEE-monotone trick: works with init = -inf (0xFF800000)
    if (v >= 0.0f) {
        __hip_atomic_fetch_max((int*)addr, __float_as_int(v),
                               __ATOMIC_RELAXED, __HIP_MEMORY_SCOPE_AGENT);
    } else {
        __hip_atomic_fetch_min((unsigned int*)addr, __float_as_uint(v),
                               __ATOMIC_RELAXED, __HIP_MEMORY_SCOPE_AGENT);
    }
}

__device__ __forceinline__ void atomicAddF(float* addr, float v) {
    __hip_atomic_fetch_add(addr, v, __ATOMIC_RELAXED, __HIP_MEMORY_SCOPE_AGENT);
}

// Build 16x32 f16 A-fragment per ISA layout.
// Lane l: row = l&15, hi = l>>4.  a[0..7] = K (hi*8 .. hi*8+7), a[8..15] = K (16+hi*8 ..).
// p must point at (row_base + kk + hi*8) in f32 memory; second run at p+16.
__device__ __forceinline__ v16h frag_from_f32(const float* __restrict__ p) {
    v16h a;
#pragma unroll
    for (int i = 0; i < 8; ++i) a[i] = (_Float16)p[i];
#pragma unroll
    for (int i = 0; i < 8; ++i) a[8 + i] = (_Float16)p[16 + i];
    return a;
}

// Same layout pattern for the B fragment (lane = column, runs along K),
// reading pre-converted f16 rows (W[n][k] row-major, padded stride).
__device__ __forceinline__ v16h frag_from_lds(const _Float16* __restrict__ p) {
    v16h a;
    h8 lo = *(const h8*)p;
    h8 hi = *(const h8*)(p + 16);
#pragma unroll
    for (int i = 0; i < 8; ++i) { a[i] = lo[i]; a[8 + i] = hi[i]; }
    return a;
}

// ---------------------------------------------------------------------------
// Kernel 1: init out / smax / denom
// ---------------------------------------------------------------------------
__global__ void gat_init_kernel(float* __restrict__ out, float* __restrict__ smax,
                                float* __restrict__ denom, int N) {
    int idx = blockIdx.x * blockDim.x + threadIdx.x;
    int total = N * D;
    if (idx < total) out[idx] = 0.0f;
    if (idx < N) { smax[idx] = -INFINITY; denom[idx] = 0.0f; }
}

// ---------------------------------------------------------------------------
// Epilogue for one 16-edge tile: bias + LeakyReLU + a_w dot + half-wave
// reduction, then store s and atomic segment-max.
// C layout: lane l holds column n = nt*16 + (l&15); VGPR g holds row
// m = g + 8*(l>>4); edge = tileBase + m.
// ---------------------------------------------------------------------------
__device__ __forceinline__ void score_epilogue(const v8f* acc, int tileBase,
                                               int r, int hi8,
                                               const float* __restrict__ sBias,
                                               const float* __restrict__ sAw,
                                               float ab,
                                               const int* __restrict__ dest,
                                               float* __restrict__ s_out,
                                               float* __restrict__ smax) {
    float partial[8];
#pragma unroll
    for (int g = 0; g < 8; ++g) partial[g] = 0.0f;
#pragma unroll
    for (int nt = 0; nt < 8; ++nt) {
        const int n = nt * 16 + r;
        const float bs = sBias[n];
        const float aw = sAw[n];
#pragma unroll
        for (int g = 0; g < 8; ++g) {
            float x = acc[nt][g] + bs;
            x = (x > 0.0f) ? x : SLOPE * x;
            partial[g] += x * aw;
        }
    }
#pragma unroll
    for (int mask = 1; mask <= 8; mask <<= 1) {
#pragma unroll
        for (int g = 0; g < 8; ++g)
            partial[g] += __shfl_xor(partial[g], mask, 32);
    }
    if (r == 0) {
#pragma unroll
        for (int g = 0; g < 8; ++g) {
            const int e = tileBase + hi8 + g;
            const float sv = partial[g] + ab;
            s_out[e] = sv;
            atomicMaxF(&smax[dest[e]], sv);
        }
    }
}

// ---------------------------------------------------------------------------
// Kernel 2: fused  s[e] = a_w . LeakyReLU(W0 M[e] + b0 + W1 M[rev[e]] + b1) + a_b
//           + atomic segment-max into smax[dest[e]]
// Persistent blocks (LDS weight conversion amortized); each wave processes
// TWO 16-edge tiles per iteration (M=32 blocking) so each LDS B-fragment load
// feeds 2 WMMAs. A compiler memory barrier at loop top pins the LDS loads
// inside the loop (prevents LICM register-hoisting -> scratch spills).
// ---------------------------------------------------------------------------
__global__ void __launch_bounds__(256)
gat_score_kernel(const float* __restrict__ M,
                 const int*   __restrict__ dest,
                 const int*   __restrict__ rev,
                 const float* __restrict__ W0, const float* __restrict__ b0,
                 const float* __restrict__ W1, const float* __restrict__ b1,
                 const float* __restrict__ a_w, const float* __restrict__ a_b,
                 float* __restrict__ s_out, float* __restrict__ smax,
                 int nPairs, int wavesTotal) {
    __shared__ __attribute__((aligned(16))) _Float16 sW0[D * LDSW];
    __shared__ __attribute__((aligned(16))) _Float16 sW1[D * LDSW];
    __shared__ float    sBias[D];
    __shared__ float    sAw[D];

    const int tid = threadIdx.x;
    // cooperative f32 -> f16 conversion of the weights into LDS (once per block)
    for (int i = tid; i < D * D; i += 256) {
        int n = i >> 7, k = i & 127;
        sW0[n * LDSW + k] = (_Float16)W0[i];
        sW1[n * LDSW + k] = (_Float16)W1[i];
    }
    for (int i = tid; i < D; i += 256) {
        sBias[i] = b0[i] + b1[i];
        sAw[i]   = a_w[i];
    }
    __syncthreads();

    const int lane = tid & 31;
    const int wave = tid >> 5;
    const int r    = lane & 15;   // A row within tile / C column / B column
    const int hi   = lane >> 4;   // half-wave selector
    const int hi8  = hi * 8;
    const float ab = a_b[0];

    const int gwave = blockIdx.x * 8 + wave;

    for (int p = gwave; p < nPairs; p += wavesTotal) {
        // Pin memory ops (esp. loop-invariant LDS B-fragment loads) inside the
        // loop: without this, LICM hoists 64 fragments into >512 VGPRs and
        // spills to scratch in the hot loop.
        asm volatile("" ::: "memory");

        const int base = p * 32;              // 32 edges per wave-iteration
        const int eA = base + r;
        const int eB = base + 16 + r;
        const int rA = rev[eA];
        const int rB = rev[eB];
        const float* pA0 = M + (size_t)eA * D + hi8;
        const float* pA1 = M + (size_t)rA * D + hi8;
        const float* pB0 = M + (size_t)eB * D + hi8;
        const float* pB1 = M + (size_t)rB * D + hi8;

        v8f accA[8], accB[8];
#pragma unroll
        for (int nt = 0; nt < 8; ++nt) {
            accA[nt] = (v8f){0,0,0,0,0,0,0,0};
            accB[nt] = (v8f){0,0,0,0,0,0,0,0};
        }

#pragma unroll
        for (int kk = 0; kk < D; kk += 32) {
            v16h aA0 = frag_from_f32(pA0 + kk);
            v16h aA1 = frag_from_f32(pA1 + kk);
            v16h aB0 = frag_from_f32(pB0 + kk);
            v16h aB1 = frag_from_f32(pB1 + kk);
#pragma unroll
            for (int nt = 0; nt < 8; ++nt) {
                const int n = nt * 16 + r;
                v16h w0 = frag_from_lds(&sW0[n * LDSW + kk + hi8]);
                v16h w1 = frag_from_lds(&sW1[n * LDSW + kk + hi8]);
                accA[nt] = __builtin_amdgcn_wmma_f32_16x16x32_f16(
                    false, aA0, false, w0, (short)0, accA[nt], false, false);
                accA[nt] = __builtin_amdgcn_wmma_f32_16x16x32_f16(
                    false, aA1, false, w1, (short)0, accA[nt], false, false);
                accB[nt] = __builtin_amdgcn_wmma_f32_16x16x32_f16(
                    false, aB0, false, w0, (short)0, accB[nt], false, false);
                accB[nt] = __builtin_amdgcn_wmma_f32_16x16x32_f16(
                    false, aB1, false, w1, (short)0, accB[nt], false, false);
            }
        }

        score_epilogue(accA, base,      r, hi8, sBias, sAw, ab, dest, s_out, smax);
        score_epilogue(accB, base + 16, r, hi8, sBias, sAw, ab, dest, s_out, smax);
    }
}

// ---------------------------------------------------------------------------
// Kernel 3: e = exp(s - smax[dest]); denom[dest] += e
// ---------------------------------------------------------------------------
__global__ void gat_exp_kernel(const float* __restrict__ s,
                               const int* __restrict__ dest,
                               const float* __restrict__ smax,
                               float* __restrict__ eexp,
                               float* __restrict__ denom, int E) {
    int e = blockIdx.x * blockDim.x + threadIdx.x;
    if (e >= E) return;
    int d = dest[e];
    float v = __expf(s[e] - smax[d]);
    eexp[e] = v;
    atomicAddF(&denom[d], v);
}

// ---------------------------------------------------------------------------
// Kernel 4: alpha = e / denom[dest]; out[dest] += alpha * M[e]
// One wave per edge; 4 f32 atomics per lane.
// ---------------------------------------------------------------------------
__global__ void __launch_bounds__(256)
gat_scatter_kernel(const float* __restrict__ M,
                   const int* __restrict__ dest,
                   const float* __restrict__ eexp,
                   const float* __restrict__ denom,
                   float* __restrict__ out,
                   float* __restrict__ alpha_out, int E) {
    const int lane = threadIdx.x & 31;
    const int wave = threadIdx.x >> 5;
    const int e = blockIdx.x * 8 + wave;
    if (e >= E) return;
    const int d = dest[e];
    const float alpha = eexp[e] / denom[d];
    if (lane == 0) alpha_out[e] = alpha;
    const float* rowM = M + (size_t)e * D;
    float* rowO = out + (size_t)d * D;
#pragma unroll
    for (int j = 0; j < 4; ++j) {
        const int c = lane + j * 32;
        atomicAddF(&rowO[c], alpha * rowM[c]);
    }
}

// ---------------------------------------------------------------------------
// Launcher
// ---------------------------------------------------------------------------
extern "C" void kernel_launch(void* const* d_in, const int* in_sizes, int n_in,
                              void* d_out, int out_size, void* d_ws, size_t ws_size,
                              hipStream_t stream) {
    const float* M    = (const float*)d_in[0];
    const int*   dest = (const int*)d_in[1];
    const int*   rev  = (const int*)d_in[2];
    // d_in[3] = dim_size (device scalar, unused: N derived from shapes)
    const float* W0   = (const float*)d_in[4];
    const float* b0   = (const float*)d_in[5];
    const float* W1   = (const float*)d_in[6];
    const float* b1   = (const float*)d_in[7];
    const float* a_w  = (const float*)d_in[8];
    const float* a_b  = (const float*)d_in[9];

    const int E = in_sizes[0] / D;          // 800000
    const int N = (out_size - E) / D;       // 50000  (out = N*D floats, then E alphas)

    float* out_mat   = (float*)d_out;
    float* alpha_out = out_mat + (size_t)N * D;

    float* ws    = (float*)d_ws;
    float* s     = ws;                      // [E]
    float* eexp  = ws + E;                  // [E]
    float* smax  = ws + 2 * (size_t)E;      // [N]
    float* denom = smax + N;                // [N]

    // K1: init
    {
        int total = N * D;
        gat_init_kernel<<<(total + 255) / 256, 256, 0, stream>>>(out_mat, smax, denom, N);
    }
    // K2: WMMA attention scores + segment max (E divisible by 32)
    {
        const int nPairs = E / 32;          // 32 edges per wave-iteration
        int blocks = 512;
        if (blocks * 8 > nPairs) blocks = (nPairs + 7) / 8;
        const int wavesTotal = blocks * 8;
        gat_score_kernel<<<blocks, 256, 0, stream>>>(M, dest, rev, W0, b0, W1, b1,
                                                     a_w, a_b, s, smax,
                                                     nPairs, wavesTotal);
    }
    // K3: exp + segment sum
    gat_exp_kernel<<<(E + 255) / 256, 256, 0, stream>>>(s, dest, smax, eexp, denom, E);
    // K4: normalize + weighted scatter into out
    gat_scatter_kernel<<<(E + 7) / 8, 256, 0, stream>>>(M, dest, eexp, denom,
                                                        out_mat, alpha_out, E);
}
